// MultiHeadCrossAttention_71210557768006
// MI455X (gfx1250) — matrix operand order repaired and checked by reading, not compiled, not run on hardware
//
#include <hip/hip_runtime.h>
#include <hip/hip_bf16.h>
#include <stdint.h>

// ---------------------------------------------------------------------------
// MultiHeadCrossAttention for MI455X (gfx1250, wave32, WMMA).
// Flash-attention design: scores never touch HBM. All matmuls via
// v_wmma_f32_16x16x32_f16 (f32 accumulate). Mask staged through LDS.
// Next-chunk global_prefetch_b8 overlaps HBM/L2 latency with WMMA work.
// Workspace layout (assumes ws_size >= 16 MiB):
//   [0,   8MiB)  kv projection, f16, (B*S) x (2*D) row-major
//   [8,  12MiB)  q  projection, f16, (B*T) x D row-major
//   [12, 16MiB)  attention output (vals), f16, (B*T) x D row-major
// ---------------------------------------------------------------------------

typedef __attribute__((ext_vector_type(16))) _Float16 v16h;
typedef __attribute__((ext_vector_type(8)))  _Float16 v8h;
typedef __attribute__((ext_vector_type(2)))  _Float16 v2h;
typedef __attribute__((ext_vector_type(8)))  float    v8f;

static constexpr int Bc = 2, Sc = 2048, Tc = 2048, Dc = 512, Hc = 8, HDc = 64;

// ---------------------------------------------------------------------------
// Generic tiled GEMM:  Out(MxN) = A(MxK) * W(KxN) + bias(N)
// Block tile 128x64, 8 waves, wave tile 32x32 (2x2 WMMA 16x16x32 f16 tiles).
// A converted to f16 in LDS; W staged transposed [n][k] so both operand
// fragments are contiguous half2 reads in this code's contiguous k order (same for A and B):
//   A frag (16x32): lane l, row = l&15, halves j: k = j + (l>>4)*16
//   B frag (32x16): lane l, col  = l&15, halves j: k = j + (l>>4)*16
// ---------------------------------------------------------------------------
template <typename AT, bool OUT_F16>
__global__ __launch_bounds__(256) void gemm_wmma_kernel(
    const AT* __restrict__ A, const float* __restrict__ W,
    const float* __restrict__ bias, void* __restrict__ Out,
    int M, int N, int K) {
  constexpr int LDA = 34;  // 32 + pad (halves)
  constexpr int LDW = 34;
  __shared__ _Float16 As[128 * LDA];
  __shared__ _Float16 Ws[64 * LDW];

  const int tid  = threadIdx.x;
  const int lane = tid & 31;
  const int wave = tid >> 5;
  const int hl   = lane >> 4;   // half-wave (0/1)
  const int ll   = lane & 15;
  const int bm   = blockIdx.y * 128;
  const int bn   = blockIdx.x * 64;
  const int wm   = (wave >> 1) * 32;  // wave M offset in block tile
  const int wn   = (wave & 1) * 32;   // wave N offset in block tile

  v8f acc[2][2] = {};

  for (int k0 = 0; k0 < K; k0 += 32) {
    // Stage A chunk 128x32 -> f16 LDS. Thread: row = tid>>1, 16 cols.
    {
      const int row = tid >> 1;
      const int c0  = (tid & 1) * 16;
      const AT* src = A + (size_t)(bm + row) * K + k0 + c0;
      _Float16* dst = &As[row * LDA + c0];
#pragma unroll
      for (int i = 0; i < 16; ++i) dst[i] = (_Float16)src[i];
      if (k0 + 32 < K) __builtin_prefetch(src + 32, 0, 3);  // next A chunk
    }
    // Stage W chunk 32x64 transposed -> Ws[n][k]. Thread: k = tid>>3, 8 cols.
    {
      const int kk = tid >> 3;
      const int n0 = (tid & 7) * 8;
      const float* src = W + (size_t)(k0 + kk) * N + bn + n0;
#pragma unroll
      for (int i = 0; i < 8; ++i) Ws[(n0 + i) * LDW + kk] = (_Float16)src[i];
      if (k0 + 32 < K) __builtin_prefetch(src + (size_t)32 * N, 0, 3);  // next W chunk
    }
    __syncthreads();

    v16h a[2], b[2];
#pragma unroll
    for (int mi = 0; mi < 2; ++mi) {
      const _Float16* base = &As[(wm + mi * 16 + ll) * LDA + hl * 16];
#pragma unroll
      for (int p = 0; p < 4; ++p) {
        v2h lo = *(const v2h*)(base + 2 * p);
        v2h hi = *(const v2h*)(base + 8 + 2 * p);
        a[mi][2 * p]     = lo.x; a[mi][2 * p + 1]     = lo.y;
        a[mi][8 + 2 * p] = hi.x; a[mi][8 + 2 * p + 1] = hi.y;
      }
    }
#pragma unroll
    for (int ni = 0; ni < 2; ++ni) {
      const _Float16* base = &Ws[(wn + ni * 16 + ll) * LDW + hl * 16];
#pragma unroll
      for (int p = 0; p < 8; ++p) {
        v2h t = *(const v2h*)(base + 2 * p);
        b[ni][2 * p] = t.x; b[ni][2 * p + 1] = t.y;
      }
    }
#pragma unroll
    for (int mi = 0; mi < 2; ++mi)
#pragma unroll
      for (int ni = 0; ni < 2; ++ni)
        acc[mi][ni] = __builtin_amdgcn_wmma_f32_16x16x32_f16(
            false, a[mi], false, b[ni], (short)0, acc[mi][ni], false, false);
    __syncthreads();
  }

  // Epilogue: +bias, store. C/D layout: row = r + 8*hl, col = ll.
#pragma unroll
  for (int mi = 0; mi < 2; ++mi)
#pragma unroll
    for (int ni = 0; ni < 2; ++ni) {
      const int gn = bn + wn + ni * 16 + ll;
      const float bv = bias[gn];
#pragma unroll
      for (int r = 0; r < 8; ++r) {
        const int gm = bm + wm + mi * 16 + r + 8 * hl;
        const float v = acc[mi][ni][r] + bv;
        if constexpr (OUT_F16)
          ((_Float16*)Out)[(size_t)gm * N + gn] = (_Float16)v;
        else
          ((float*)Out)[(size_t)gm * N + gn] = v;
      }
    }
}

// ---------------------------------------------------------------------------
// Flash attention. Grid: B*H*(T/128) blocks of 256 threads (8 waves).
// Wave w owns query rows [t0 + 16w, t0 + 16w + 16). S consumed in 64-chunks.
// KVb rows: (b*S+s) x (2*D); head h: K = cols [h*128, h*128+64), V = +64.
// Mask chunk (128 t-rows x 64 s-cols) staged through LDS (coalesced b128).
// ---------------------------------------------------------------------------
__global__ __launch_bounds__(256) void flash_attn_kernel(
    const _Float16* __restrict__ Qb, const _Float16* __restrict__ KVb,
    const uint8_t* __restrict__ mask, _Float16* __restrict__ Vals) {
  constexpr int LDK = 72;                 // 64 + pad, keeps 16B alignment
  __shared__ _Float16 Ks[64 * LDK];       // [s][hd]
  __shared__ _Float16 Vs[64 * LDK];       // [hd][s]  (transposed)
  __shared__ _Float16 Ps[128 * LDK];      // per-wave 16x64 softmax tiles
  __shared__ uint8_t  Ms[128 * 64];       // mask chunk [t_row][s_col]

  const int tid  = threadIdx.x;
  const int lane = tid & 31;
  const int wave = tid >> 5;
  const int hl   = lane >> 4;
  const int ll   = lane & 15;

  const int tTiles = Tc / 128;
  const int b  = blockIdx.x / (Hc * tTiles);
  const int h  = (blockIdx.x / tTiles) % Hc;
  const int t0 = (blockIdx.x % tTiles) * 128;

  // Q A-fragments (16 rows x 64 head-dim = 2 fragments), kept in registers.
  v16h aq[2];
  {
    const int trow = t0 + wave * 16 + ll;
    const _Float16* qrow = Qb + (size_t)(b * Tc + trow) * Dc + h * HDc;
#pragma unroll
    for (int kc = 0; kc < 2; ++kc) {
      const _Float16* base = qrow + kc * 32 + hl * 16;
#pragma unroll
      for (int p = 0; p < 4; ++p) {
        v2h lo = *(const v2h*)(base + 2 * p);
        v2h hi = *(const v2h*)(base + 8 + 2 * p);
        aq[kc][2 * p]     = lo.x; aq[kc][2 * p + 1]     = lo.y;
        aq[kc][8 + 2 * p] = hi.x; aq[kc][8 + 2 * p + 1] = hi.y;
      }
    }
  }

  float run_max[8], run_sum[8];
  v8f o[4] = {};
#pragma unroll
  for (int r = 0; r < 8; ++r) { run_max[r] = -3.0e38f; run_sum[r] = 0.0f; }
  const float scale = 0.125f;  // 1/sqrt(64)
  const uint8_t* maskTile = mask + (size_t)b * Tc * Sc + (size_t)t0 * Sc;

  for (int s0 = 0; s0 < Sc; s0 += 64) {
    __syncthreads();  // previous chunk fully consumed before overwrite
    // ---- cooperative stage: K, V (transposed), mask ----
    {
      const int s  = tid >> 2;
      const int c0 = (tid & 3) * 16;
      const _Float16* kv = KVb + (size_t)(b * Sc + s0 + s) * (2 * Dc) + h * 2 * HDc;
      *(v8h*)&Ks[s * LDK + c0]     = *(const v8h*)(kv + c0);
      *(v8h*)&Ks[s * LDK + c0 + 8] = *(const v8h*)(kv + c0 + 8);
      const _Float16* vv = kv + HDc;
#pragma unroll
      for (int i = 0; i < 16; ++i) Vs[(c0 + i) * LDK + s] = vv[c0 + i];
      if (s0 + 64 < Sc)  // prefetch next chunk's KV rows (K and V halves)
        __builtin_prefetch(kv + (size_t)64 * (2 * Dc) + c0 * 2, 0, 3);
    }
    {
      const int mr = tid >> 1;             // mask row within tile (0..127)
      const int mc = (tid & 1) * 32;       // 32-byte half of the 64-col chunk
      const uint8_t* msrc = maskTile + (size_t)mr * Sc + s0 + mc;
      *(uint32_t*)(&Ms[mr * 64 + mc] + 0)      = ((const uint32_t*)msrc)[0];
      *(uint4*)(&Ms[mr * 64 + mc] + 4 - 4)     = *(const uint4*)msrc;       // 16B
      *(uint4*)(&Ms[mr * 64 + mc + 16])        = *(const uint4*)(msrc + 16);
      if (s0 + 64 < Sc) __builtin_prefetch(msrc + 64, 0, 3);
    }
    __syncthreads();

    // ---- scores = Q K^T, mask + scale ----
    float sc[4][8];
#pragma unroll
    for (int ni = 0; ni < 4; ++ni) {
      v8f accs = {};
#pragma unroll
      for (int kc = 0; kc < 2; ++kc) {
        v16h kb;
        const _Float16* base = &Ks[(ni * 16 + ll) * LDK + kc * 32 + hl * 16];
#pragma unroll
        for (int p = 0; p < 8; ++p) {
          v2h t = *(const v2h*)(base + 2 * p);
          kb[2 * p] = t.x; kb[2 * p + 1] = t.y;
        }
        accs = __builtin_amdgcn_wmma_f32_16x16x32_f16(
            false, aq[kc], false, kb, (short)0, accs, false, false);
      }
      const uint8_t* mrow = &Ms[(wave * 16 + 8 * hl) * 64 + ni * 16 + ll];
#pragma unroll
      for (int r = 0; r < 8; ++r)
        sc[ni][r] = mrow[r * 64] ? accs[r] * scale : -1.0e9f;
    }

    // ---- online softmax (row lives in one 16-lane half) ----
#pragma unroll
    for (int r = 0; r < 8; ++r) {
      float cm = fmaxf(fmaxf(sc[0][r], sc[1][r]), fmaxf(sc[2][r], sc[3][r]));
#pragma unroll
      for (int off = 8; off >= 1; off >>= 1)
        cm = fmaxf(cm, __shfl_xor(cm, off, 32));
      const float nm   = fmaxf(run_max[r], cm);
      const float corr = __expf(run_max[r] - nm);
      run_max[r] = nm;
      float cs = 0.0f;
#pragma unroll
      for (int ni = 0; ni < 4; ++ni) {
        sc[ni][r] = __expf(sc[ni][r] - nm);
        cs += sc[ni][r];
      }
#pragma unroll
      for (int off = 8; off >= 1; off >>= 1)
        cs += __shfl_xor(cs, off, 32);
      run_sum[r] = run_sum[r] * corr + cs;
#pragma unroll
      for (int ni = 0; ni < 4; ++ni) o[ni][r] *= corr;
    }

    // ---- P through wave-private LDS (C-layout -> A-fragment layout) ----
#pragma unroll
    for (int ni = 0; ni < 4; ++ni)
#pragma unroll
      for (int r = 0; r < 8; ++r)
        Ps[(wave * 16 + r + 8 * hl) * LDK + ni * 16 + ll] = (_Float16)sc[ni][r];

    v16h ap[2];
#pragma unroll
    for (int kc = 0; kc < 2; ++kc) {
      const _Float16* base = &Ps[(wave * 16 + ll) * LDK + kc * 32 + hl * 16];
#pragma unroll
      for (int p = 0; p < 4; ++p) {
        v2h lo = *(const v2h*)(base + 2 * p);
        v2h hi = *(const v2h*)(base + 8 + 2 * p);
        ap[kc][2 * p]     = lo.x; ap[kc][2 * p + 1]     = lo.y;
        ap[kc][8 + 2 * p] = hi.x; ap[kc][8 + 2 * p + 1] = hi.y;
      }
    }

    // ---- O += P V ----
#pragma unroll
    for (int ni = 0; ni < 4; ++ni)
#pragma unroll
      for (int kc = 0; kc < 2; ++kc) {
        v16h vb;
        const _Float16* base = &Vs[(ni * 16 + ll) * LDK + kc * 32 + hl * 16];
#pragma unroll
        for (int p = 0; p < 8; ++p) {
          v2h t = *(const v2h*)(base + 2 * p);
          vb[2 * p] = t.x; vb[2 * p + 1] = t.y;
        }
        o[ni] = __builtin_amdgcn_wmma_f32_16x16x32_f16(
            false, ap[kc], false, vb, (short)0, o[ni], false, false);
      }
  }

  // ---- normalize + store vals (f16, (B*T) x D row-major) ----
#pragma unroll
  for (int ni = 0; ni < 4; ++ni)
#pragma unroll
    for (int r = 0; r < 8; ++r) {
      const int trow = t0 + wave * 16 + r + 8 * hl;
      const int hd   = ni * 16 + ll;
      const float v  = o[ni][r] / run_sum[r];
      Vals[(size_t)(b * Tc + trow) * Dc + h * HDc + hd] = (_Float16)v;
    }
}

// ---------------------------------------------------------------------------
extern "C" void kernel_launch(void* const* d_in, const int* in_sizes, int n_in,
                              void* d_out, int out_size, void* d_ws, size_t ws_size,
                              hipStream_t stream) {
  const float*   x    = (const float*)d_in[0];
  const float*   y    = (const float*)d_in[1];
  const uint8_t* mask = (const uint8_t*)d_in[2];  // numpy bool = 1 byte
  const float*   kv_w = (const float*)d_in[3];
  const float*   kv_b = (const float*)d_in[4];
  const float*   q_w  = (const float*)d_in[5];
  const float*   q_b  = (const float*)d_in[6];
  const float*   o_w  = (const float*)d_in[7];
  const float*   o_b  = (const float*)d_in[8];
  float* out = (float*)d_out;

  char* ws = (char*)d_ws;
  _Float16* ws_kv = (_Float16*)ws;                          // 4096 x 1024 f16 (8 MiB)
  _Float16* ws_q  = (_Float16*)(ws + ((size_t)8  << 20));   // 4096 x 512  f16 (4 MiB)
  _Float16* ws_v  = (_Float16*)(ws + ((size_t)12 << 20));   // 4096 x 512  f16 (4 MiB)

  const int M  = Bc * Sc;       // 4096 (== B*T too)
  const dim3 blk(256);

  // kv = x @ kv_w + kv_b   -> f16
  gemm_wmma_kernel<float, true>
      <<<dim3((2 * Dc) / 64, M / 128), blk, 0, stream>>>(x, kv_w, kv_b, ws_kv, M, 2 * Dc, Dc);
  // q = y @ q_w + q_b      -> f16
  gemm_wmma_kernel<float, true>
      <<<dim3(Dc / 64, M / 128), blk, 0, stream>>>(y, q_w, q_b, ws_q, M, Dc, Dc);
  // flash attention        -> vals f16
  flash_attn_kernel<<<dim3(Bc * Hc * (Tc / 128)), blk, 0, stream>>>(ws_q, ws_kv, mask, ws_v);
  // out = vals @ o_w + o_b -> f32
  gemm_wmma_kernel<_Float16, false>
      <<<dim3(Dc / 64, M / 128), blk, 0, stream>>>(ws_v, o_w, o_b, out, M, Dc, Dc);
}
